// FuncSelfAttention_89790586290689
// MI455X (gfx1250) — matrix operand order repaired and checked by reading, not compiled, not run on hardware
//
#include <hip/hip_runtime.h>
#include <hip/hip_bf16.h>
#include <stdint.h>

#define B_   4
#define S_   256
#define E_   256
#define H_   8
#define D_   32
#define HW_  256
#define T_   (B_*S_)     // 1024 tokens
#define E3_  (3*E_)      // 768
#define F_   (D_*HW_)    // 8192 features per head

typedef __attribute__((ext_vector_type(16))) __bf16 v16bf;
typedef __attribute__((ext_vector_type(8)))  float  v8f;
typedef __attribute__((ext_vector_type(4)))  unsigned int u32x4;
typedef __attribute__((ext_vector_type(8)))  int i32x8;
typedef __attribute__((ext_vector_type(4)))  int i32x4;

union V16 { uint32_t u[8]; v16bf v; };

static __device__ inline __bf16 f2bf(float f) {
  union { float f; uint32_t u; } x; x.f = f;
  uint32_t r = x.u + 0x7FFFu + ((x.u >> 16) & 1u);   // RNE
  uint16_t h = (uint16_t)(r >> 16);
  __bf16 b; __builtin_memcpy(&b, &h, 2); return b;
}

static __device__ inline v8f wmma_bf16(v16bf a, v16bf b, v8f c) {
  return __builtin_amdgcn_wmma_f32_16x16x32_bf16(false, a, false, b,
                                                 (short)0, c, false, false);
}

// A tile 16x32 bf16: rows are M (stride rowStride), K contiguous.
static __device__ inline v16bf loadA(const __bf16* src, size_t rowStride, int k0) {
  int lane = threadIdx.x & 31;
  int m = lane & 15;
  int kh = (lane >> 4) << 4;
  const __bf16* p = src + (size_t)m * rowStride + (size_t)(k0 + kh);
  V16 r;
#pragma unroll
  for (int i = 0; i < 4; ++i) r.u[i]     = *(const uint32_t*)(p + 2*i);
#pragma unroll
  for (int i = 0; i < 4; ++i) r.u[4 + i] = *(const uint32_t*)(p + 8 + 2*i);
  return r.v;
}

// B tile 32x16 bf16 where MEMORY holds B transposed: src is [n][k] row-major.
static __device__ inline v16bf loadBt(const __bf16* src, size_t rowStride, int k0) {
  int lane = threadIdx.x & 31;
  int n = lane & 15;
  int kb = (lane >> 4) << 4;
  const __bf16* p = src + (size_t)n * rowStride + (size_t)(k0 + kb);
  V16 r;
#pragma unroll
  for (int i = 0; i < 8; ++i) r.u[i] = *(const uint32_t*)(p + 2*i);
  return r.v;
}

// B tile 32x16 bf16 from an LDS panel laid out [n(256 rows)][k(32)].
static __device__ inline v16bf loadB_lds(const __bf16* p, int n0) {
  int lane = threadIdx.x & 31;
  int row = n0 + (lane & 15);
  int kb = (lane >> 4) << 4;
  const __bf16* q = p + row * 32 + kb;
  V16 r;
#pragma unroll
  for (int i = 0; i < 8; ++i) r.u[i] = *(const uint32_t*)(q + 2*i);
  return r.v;
}

static __device__ inline uint32_t ldsoff(const void* p) {
  return (uint32_t)(uintptr_t)p;   // LDS generic addr: low 32 bits = offset
}

// Tensor Data Mover: 2D tile load (bf16 elements) global -> LDS.
// D# per CDNA5 ISA 8.3/8.4: group0 = {count/type, lds_addr, global_addr},
// group1 = {data_size, tensor dims, tile dims, dim0 stride}.
// This toolchain exposes the 6-arg builtin:
//   (u32x4 g0, i32x8 g1, i32x4 g2, i32x4 g3, i32x8 g4, i32 cpol)
static __device__ inline void tdm_load_2d(uint32_t lds_byte_off, const void* gptr,
                                          uint32_t tile_d0, uint32_t tile_d1,
                                          uint32_t tensor_d0, uint32_t tensor_d1,
                                          uint32_t stride0) {
  uint64_t ga = (uint64_t)gptr;
  u32x4 g0;
  g0[0] = 1u;                                        // count=1, user mode
  g0[1] = lds_byte_off;                              // lds_addr
  g0[2] = (uint32_t)ga;                              // global_addr[31:0]
  g0[3] = (uint32_t)((ga >> 32) & 0x01FFFFFFu)       // global_addr[56:32]
        | (2u << 30);                                // type=2 (image)
  i32x8 g1;
  g1[0] = (int)(1u << 16);                           // data_size=1 -> 2 bytes
  g1[1] = (int)((tensor_d0 & 0xFFFFu) << 16);        // tensor_dim0[15:0]
  g1[2] = (int)((tensor_d0 >> 16) | ((tensor_d1 & 0xFFFFu) << 16));
  g1[3] = (int)((tensor_d1 >> 16) | (tile_d0 << 16));// tile_dim0
  g1[4] = (int)(tile_d1 & 0xFFFFu);                  // tile_dim1 (tile_dim2=0)
  g1[5] = (int)stride0;                              // tensor_dim0_stride lo
  g1[6] = 0;
  g1[7] = 0;
  i32x4 z4 = {0, 0, 0, 0};
  i32x8 z8 = {0, 0, 0, 0, 0, 0, 0, 0};
  __builtin_amdgcn_tensor_load_to_lds(g0, g1, z4, z4, z8, 0);
}

// ---------------- K0a: weights fp32 -> bf16 ----------------
__global__ void k_convert_weights(const float* __restrict__ wqkv,
                                  const float* __restrict__ wout,
                                  __bf16* __restrict__ wqkv_bf,
                                  __bf16* __restrict__ wout_bf) {
  int i = blockIdx.x * blockDim.x + threadIdx.x;
  if (i < E3_ * E_) wqkv_bf[i] = f2bf(wqkv[i]);
  if (i < E_ * E_)  wout_bf[i] = f2bf(wout[i]);
}

// ---------------- K0b: seq [t][c][xy] fp32 -> xt [t][xy][c] bf16 -------------
__global__ void k_transpose_seq(const float* __restrict__ seq,
                                __bf16* __restrict__ xt) {
  size_t i = (size_t)blockIdx.x * blockDim.x + threadIdx.x;
  if (i >= (size_t)T_ * E_ * HW_) return;
  int xy = (int)(i & (HW_ - 1));
  int c  = (int)((i >> 8) & (E_ - 1));
  size_t t = i >> 16;
  xt[t * (size_t)(E_ * HW_) + (size_t)xy * E_ + c] = f2bf(seq[i]);
}

// ---------------- K1: QKV projection with TDM-staged X panels ----------------
// grid (T_, 12) x 256. Block: 64 M-rows x 256 N. X K-panels (32c x 256xy bf16,
// 16KB) stream into LDS via tensor_load_to_lds, double-buffered on TENSORcnt.
// V tiles additionally stored transposed into vT[b,h][f][k] for k_sa.
__global__ void __launch_bounds__(256) k_qkv(const __bf16* __restrict__ wqkv_bf,
                                             const __bf16* __restrict__ xt,
                                             __bf16* __restrict__ qkv,
                                             __bf16* __restrict__ vT) {
  __shared__ __bf16 panel[2][32 * HW_];   // 2 x 16KB
  int t = blockIdx.x;
  int b = t >> 8, s = t & 255;
  int wave = threadIdx.x >> 5;
  int lane = threadIdx.x & 31;
  int mt  = blockIdx.y * 4 + (wave >> 1);  // 0..47
  int nt0 = (wave & 1) * 8;                // 8 n-tiles of 16
  const __bf16* A = wqkv_bf + (size_t)(mt * 16) * E_;
  const __bf16* xbase = xt + (size_t)t * (E_ * HW_);

  if (wave == 0)
    tdm_load_2d(ldsoff(&panel[0][0]), xbase, 32, HW_, E_, HW_, E_);

  v8f acc[8] = {};
  for (int step = 0; step < 8; ++step) {
    int buf = step & 1;
    __syncthreads();                       // consumers of panel[buf^1] done
    if (wave == 0) {
      if (step < 7) {
        tdm_load_2d(ldsoff(&panel[buf ^ 1][0]), xbase + (step + 1) * 32,
                    32, HW_, E_, HW_, E_);
        __builtin_amdgcn_s_wait_tensorcnt(1);   // in-order: panel[buf] ready
      } else {
        __builtin_amdgcn_s_wait_tensorcnt(0);
      }
    }
    __syncthreads();                       // publish panel[buf] to all waves
    v16bf a = loadA(A, E_, step * 32);
#pragma unroll
    for (int j = 0; j < 8; ++j)
      acc[j] = wmma_bf16(a, loadB_lds(&panel[buf][0], (nt0 + j) * 16), acc[j]);
  }

  int row0 = (lane >> 4) << 3;
  int col  = lane & 15;
  int head = mt / 6;
  int remt = mt % 6;                        // tiles 4,5 within a head are V
#pragma unroll
  for (int j = 0; j < 8; ++j) {
    int xy = (nt0 + j) * 16 + col;
    __bf16* dst = qkv + (size_t)t * (E3_ * HW_) + (size_t)(mt * 16) * HW_ + xy;
#pragma unroll
    for (int jj = 0; jj < 8; ++jj)
      dst[(size_t)(row0 + jj) * HW_] = f2bf(acc[j][jj]);
    if (remt >= 4) {
#pragma unroll
      for (int jj = 0; jj < 8; ++jj) {
        int dd = (remt - 4) * 16 + row0 + jj;
        int f  = dd * HW_ + xy;
        vT[((size_t)(b * H_ + head) * F_ + f) * S_ + s] = f2bf(acc[j][jj]);
      }
    }
  }
}

// ---------------- K2: scores = Q K^T * scale, softmax, attn bf16 -------------
// grid (B_*H_, S_/32) x 256; block computes 32 q-rows x 256 k-cols.
__global__ void __launch_bounds__(256) k_scores(const __bf16* __restrict__ qkv,
                                                __bf16* __restrict__ attn) {
  __shared__ float sc[32][256];
  __shared__ float red[32][8];
  int bh = blockIdx.x;
  int b = bh >> 3, h = bh & 7;
  int q0 = blockIdx.y * 32;
  int wave = threadIdx.x >> 5;
  int qt  = wave >> 2;
  int kt0 = (wave & 3) * 4;

  const size_t tokStride = (size_t)E3_ * HW_;
  const __bf16* Qm = qkv + (size_t)(b * S_ + q0 + qt * 16) * tokStride
                         + (size_t)(h * 96) * HW_;
  const __bf16* Km = qkv + (size_t)(b * S_) * tokStride
                         + (size_t)(h * 96 + 32) * HW_;
  v8f acc[4] = {};
  for (int ks = 0; ks < F_; ks += 32) {
    v16bf a = loadA(Qm, tokStride, ks);
#pragma unroll
    for (int j = 0; j < 4; ++j) {
      v16bf bm = loadBt(Km + (size_t)((kt0 + j) * 16) * tokStride, tokStride, ks);
      acc[j] = wmma_bf16(a, bm, acc[j]);
    }
  }
  const float scale = 0.011048543456039804f;   // 1/sqrt(32*16*16)
  int lane = threadIdx.x & 31;
  int row0 = (lane >> 4) << 3;
  int col  = lane & 15;
#pragma unroll
  for (int kt = 0; kt < 4; ++kt)
#pragma unroll
    for (int j = 0; j < 8; ++j)
      sc[qt * 16 + row0 + j][(kt0 + kt) * 16 + col] = acc[kt][j] * scale;
  __syncthreads();

  int r   = threadIdx.x >> 3;
  int seg = threadIdx.x & 7;
  float mx = -1e30f;
  for (int i = 0; i < 32; ++i) mx = fmaxf(mx, sc[r][seg * 32 + i]);
  red[r][seg] = mx;
  __syncthreads();
  mx = red[r][0];
  for (int i = 1; i < 8; ++i) mx = fmaxf(mx, red[r][i]);
  float sum = 0.f;
  for (int i = 0; i < 32; ++i) {
    float e = __expf(sc[r][seg * 32 + i] - mx);
    sc[r][seg * 32 + i] = e;
    sum += e;
  }
  __syncthreads();
  red[r][seg] = sum;
  __syncthreads();
  float tot = 0.f;
  for (int i = 0; i < 8; ++i) tot += red[r][i];
  float inv = 1.0f / tot;
  __bf16* dst = attn + ((size_t)bh * S_ + (size_t)(q0 + r)) * S_;
  for (int i = 0; i < 32; ++i) dst[seg * 32 + i] = f2bf(sc[r][seg * 32 + i] * inv);
}

// ---------------- K3a: sa = attn @ V from vT (contiguous B), saT out ---------
// grid (B_*H_, 4, 32) x 256; block: 64 q x 256 features; wave: 8 f-tiles.
__global__ void __launch_bounds__(256) k_sa(const __bf16* __restrict__ attn,
                                            const __bf16* __restrict__ vT,
                                            __bf16* __restrict__ saT) {
  int bh = blockIdx.x; int b = bh >> 3, h = bh & 7;
  int q0 = blockIdx.y * 64;
  int f0 = blockIdx.z * 256;
  int wave = threadIdx.x >> 5;
  int lane = threadIdx.x & 31;
  int qt  = wave >> 1;           // 0..3
  int ft0 = (wave & 1) * 8;      // 8 f-tiles of 16

  const __bf16* Am = attn + ((size_t)bh * S_ + q0 + qt * 16) * S_;
  const __bf16* Vm = vT + ((size_t)bh * F_ + f0) * S_;

  v8f acc[8] = {};
#pragma unroll
  for (int ks = 0; ks < S_; ks += 32) {
    v16bf a = loadA(Am, S_, ks);
#pragma unroll
    for (int j = 0; j < 8; ++j)
      acc[j] = wmma_bf16(a, loadBt(Vm + (size_t)((ft0 + j) * 16) * S_, S_, ks),
                         acc[j]);
  }
  int row0 = (lane >> 4) << 3;
  int col  = lane & 15;
#pragma unroll
  for (int j = 0; j < 8; ++j) {
    int f  = f0 + (ft0 + j) * 16 + col;
    int dd = f >> 8, xy = f & 255;
    __bf16* dst = saT + (size_t)(b * S_ + q0 + qt * 16 + row0) * (E_ * HW_)
                      + (size_t)xy * E_ + (size_t)(h * 32 + dd);
#pragma unroll
    for (int jj = 0; jj < 8; ++jj)
      dst[(size_t)jj * (E_ * HW_)] = f2bf(acc[j][jj]);
  }
}

// ---------------- K3b: out = Wout @ sa_t + bias (f32 out) --------------------
__global__ void __launch_bounds__(256) k_out(const __bf16* __restrict__ wout_bf,
                                             const __bf16* __restrict__ saT,
                                             const float* __restrict__ bias,
                                             float* __restrict__ out) {
  int t    = blockIdx.x;
  int wave = threadIdx.x >> 5;
  int tile = blockIdx.y * 8 + wave;
  int mt = tile >> 4, nt = tile & 15;
  const __bf16* A  = wout_bf + (size_t)(mt * 16) * E_;
  const __bf16* Bm = saT + (size_t)t * (E_ * HW_) + (size_t)(nt * 16) * E_;
  v8f acc = {};
#pragma unroll
  for (int ks = 0; ks < E_; ks += 32)
    acc = wmma_bf16(loadA(A, E_, ks), loadBt(Bm, E_, ks), acc);

  int lane = threadIdx.x & 31;
  int row0 = (lane >> 4) << 3;
  int col  = lane & 15;
  float* dst = out + (size_t)t * (E_ * HW_)
                   + (size_t)(mt * 16) * HW_ + (size_t)(nt * 16 + col);
#pragma unroll
  for (int j = 0; j < 8; ++j)
    dst[(size_t)(row0 + j) * HW_] = acc[j] + bias[mt * 16 + row0 + j];
}

extern "C" void kernel_launch(void* const* d_in, const int* in_sizes, int n_in,
                              void* d_out, int out_size, void* d_ws, size_t ws_size,
                              hipStream_t stream) {
  const float* seq  = (const float*)d_in[0];
  const float* wqkv = (const float*)d_in[1];
  const float* wout = (const float*)d_in[2];
  const float* bout = (const float*)d_in[3];
  float* out = (float*)d_out;

  char* ws = (char*)d_ws;
  size_t off = 0;
  auto take = [&](size_t bytes) -> void* {
    void* p = ws + off;
    off = (off + bytes + 255) & ~(size_t)255;
    return p;
  };
  __bf16* wqkv_bf = (__bf16*)take((size_t)E3_ * E_ * 2);
  __bf16* wout_bf = (__bf16*)take((size_t)E_ * E_ * 2);
  __bf16* xt      = (__bf16*)take((size_t)T_ * E_ * HW_ * 2);        // [t][xy][c]
  __bf16* qkv     = (__bf16*)take((size_t)T_ * E3_ * HW_ * 2);       // [t][o][xy]
  __bf16* attn    = (__bf16*)take((size_t)B_ * H_ * S_ * S_ * 2);
  __bf16* saT     = (__bf16*)take((size_t)T_ * E_ * HW_ * 2);        // [t][xy][c]
  __bf16* vT      = (__bf16*)take((size_t)B_ * H_ * F_ * S_ * 2);    // [bh][f][k]

  k_convert_weights<<<(E3_ * E_ + 255) / 256, 256, 0, stream>>>(
      wqkv, wout, wqkv_bf, wout_bf);
  k_transpose_seq<<<(int)(((size_t)T_ * E_ * HW_ + 255) / 256), 256, 0, stream>>>(
      seq, xt);
  k_qkv<<<dim3(T_, 12), 256, 0, stream>>>(wqkv_bf, xt, qkv, vT);
  k_scores<<<dim3(B_ * H_, S_ / 32), 256, 0, stream>>>(qkv, attn);
  k_sa<<<dim3(B_ * H_, 4, 32), 256, 0, stream>>>(attn, vT, saT);
  k_out<<<dim3(T_, 32), 256, 0, stream>>>(wout_bf, saT, bout, out);
}
